// Monet_48730698941115
// MI455X (gfx1250) — matrix operand order repaired and checked by reading, not compiled, not run on hardware
//
#include <hip/hip_runtime.h>
#include <hip/hip_bf16.h>

// ---------------------------------------------------------------------------
// MoNet (GMMConv x4 + BN) -> BiLSTM JumpingKnowledge -> pool -> MLP -> LN
// MI455X / gfx1250: all dense GEMMs go through v_wmma_f32_16x16x32_f16
// (f16 in, f32 accum). 128x128 block tile, double-buffered LDS, all weights
// pre-transposed to [N][K] so staging is pure b128 vector traffic. All 8
// B fragments are preloaded per K-step so the 8 WMMAs issue back-to-back
// behind a single s_wait_dscnt.
// ---------------------------------------------------------------------------

typedef __attribute__((ext_vector_type(16))) _Float16     v16h;
typedef __attribute__((ext_vector_type(8)))  float        v8f;
typedef __attribute__((ext_vector_type(4)))  unsigned int v4u;

#define kN    30000
#define kE    480000
#define kC    128          // F == C == 128
#define kK    8
#define kD    2
#define kL    4
#define kH    256          // LSTM hidden
#define kH4   1024         // 4*H
#define kKC   1024         // K*C
#define kNHID 512
#define kNOUT 256
#define kB    1024
#define kT    5            // L+1 JK steps

union Frag16 { v16h v; v4u q[2]; };

// ---------------------------------------------------------------------------
// WMMA GEMM:  C[M,N](f32) = A[M,K](f16,row) x Bt[N,K](f16,row = B^T)
//             [+C_prev][+bias][relu]
// Requires K%32==0, N%128==0 (true for all call sites: K in {128,256,512},
// N in {128,256,512,1024}). Block tile 128x128: 8 wave32 waves, each wave
// owns a 16-row strip x 8 column tiles (8 v8f accumulators, 8 WMMAs/K-step).
// ---------------------------------------------------------------------------
__global__ __launch_bounds__(256) void gemm_f16_wmma(
    const _Float16* __restrict__ A, const _Float16* __restrict__ Bt,
    float* __restrict__ Cmat, const float* __restrict__ bias,
    int M, int Kd, int Nd, int accum, int relu)
{
  __shared__ _Float16 sA[2][128][32];   // 2 x 8 KB
  __shared__ _Float16 sB[2][128][32];   // 2 x 8 KB  (rows = N cols of B)

  const int tid  = threadIdx.x;
  const int wave = tid >> 5;         // 0..7 (wave32)
  const int lane = tid & 31;
  const int half = lane >> 4;
  const int l16  = lane & 15;
  const int bm   = blockIdx.x * 128;
  const int bn   = blockIdx.y * 128;

  // staging assignment: thread t covers 16 contiguous halves of one row
  const int row   = tid >> 1;          // 0..127
  const int chunk = (tid & 1) * 16;    // 0 / 16 (halves)
  const _Float16* aRow = A  + (size_t)(bm + row) * Kd + chunk;
  const _Float16* bRow = Bt + (size_t)(bn + row) * Kd + chunk;
  const bool aOk = (bm + row) < M;

  v8f acc[8] = {};

  { // stage K-tile 0
    v4u a0 = {}, a1 = {};
    if (aOk) { const v4u* p = (const v4u*)aRow; a0 = p[0]; a1 = p[1]; }
    const v4u* q = (const v4u*)bRow;
    *(v4u*)&sA[0][row][chunk]     = a0;
    *(v4u*)&sA[0][row][chunk + 8] = a1;
    *(v4u*)&sB[0][row][chunk]     = q[0];
    *(v4u*)&sB[0][row][chunk + 8] = q[1];
  }
  __syncthreads();

  int cur = 0;
  for (int k0 = 0; k0 < Kd; k0 += 32) {
    const int nxt = k0 + 32;
    // issue next-tile global loads first (s_wait lands at the LDS stores,
    // after the WMMAs have been issued from the current buffer)
    v4u a0 = {}, a1 = {}, b0 = {}, b1 = {};
    if (nxt < Kd) {
      if (aOk) {
        const v4u* p = (const v4u*)(aRow + nxt); a0 = p[0]; a1 = p[1];
        if (nxt + 32 < Kd) __builtin_prefetch(aRow + nxt + 32, 0, 1);
      }
      const v4u* q = (const v4u*)(bRow + nxt); b0 = q[0]; b1 = q[1];
    }

    // A fragment: lane = row (wave*16+l16); K = half*16..+15
    Frag16 a;
    a.q[0] = *(const v4u*)&sA[cur][wave * 16 + l16][half * 16];
    a.q[1] = *(const v4u*)&sA[cur][wave * 16 + l16][half * 16 + 8];

    // preload ALL 8 B fragments (lane = col c*16+l16; K = half*16..+15),
    // then fire the 8 WMMAs back-to-back behind one dscnt wait
    Frag16 bf[8];
    #pragma unroll
    for (int c = 0; c < 8; ++c) {
      const v4u* bp = (const v4u*)&sB[cur][c * 16 + l16][half * 16];
      bf[c].q[0] = bp[0]; bf[c].q[1] = bp[1];
    }
    #pragma unroll
    for (int c = 0; c < 8; ++c) {
      acc[c] = __builtin_amdgcn_wmma_f32_16x16x32_f16(
                   false, a.v, false, bf[c].v, (short)0, acc[c], false, false);
    }

    if (nxt < Kd) {
      *(v4u*)&sA[cur ^ 1][row][chunk]     = a0;
      *(v4u*)&sA[cur ^ 1][row][chunk + 8] = a1;
      *(v4u*)&sB[cur ^ 1][row][chunk]     = b0;
      *(v4u*)&sB[cur ^ 1][row][chunk + 8] = b1;
    }
    __syncthreads();
    cur ^= 1;
  }

  #pragma unroll
  for (int c = 0; c < 8; ++c) {
    const int col = bn + c * 16 + l16;
    #pragma unroll
    for (int r = 0; r < 8; ++r) {
      const int rw = bm + wave * 16 + half * 8 + r;   // C/D VGPR layout
      if (rw < M) {
        const size_t idx = (size_t)rw * Nd + col;
        float v = acc[c][r];
        if (accum) v += Cmat[idx];
        if (bias)  v += bias[col];
        if (relu)  v = fmaxf(v, 0.f);
        Cmat[idx] = v;
      }
    }
  }
}

// ------------------------------ helpers ------------------------------------
__device__ __forceinline__ float sigmf(float x) { return 1.f / (1.f + __expf(-x)); }

__global__ void k_f32_to_f16(const float* __restrict__ in, _Float16* __restrict__ out, int n) {
  int i = blockIdx.x * blockDim.x + threadIdx.x;
  if (i < n) out[i] = (_Float16)in[i];
}

// out[Cc x R] = in[R x Cc]^T, f32 -> f16
__global__ void k_transpose_f16(const float* __restrict__ in, _Float16* __restrict__ out,
                                int R, int Cc) {
  int i = blockIdx.x * blockDim.x + threadIdx.x;
  if (i < R * Cc) {
    int r = i / Cc, c = i % Cc;
    out[(size_t)c * R + r] = (_Float16)in[i];
  }
}

__global__ void k_bias_sum(const float* __restrict__ a, const float* __restrict__ b,
                           float* __restrict__ o, int n) {
  int i = blockIdx.x * blockDim.x + threadIdx.x;
  if (i < n) o[i] = a[i] + b[i];
}

__global__ void k_deg(const int* __restrict__ dst, float* __restrict__ deg, int n) {
  int i = blockIdx.x * blockDim.x + threadIdx.x;
  if (i < n) atomicAdd(&deg[dst[i]], 1.f);
}

// Gaussian kernel weights per edge: w[e,k] = exp(sum_d -0.5 diff^2/(eps+sig^2))
__global__ void k_edge_w(const float* __restrict__ eattr, const float* __restrict__ mu,
                         const float* __restrict__ sig, float* __restrict__ w, int n) {
  int e = blockIdx.x * blockDim.x + threadIdx.x;
  if (e >= n) return;
  float u0 = eattr[e * kD + 0], u1 = eattr[e * kD + 1];
  #pragma unroll
  for (int k = 0; k < kK; ++k) {
    float d0 = u0 - mu[k * kD + 0], d1 = u1 - mu[k * kD + 1];
    float s0 = sig[k * kD + 0], s1 = sig[k * kD + 1];
    float s = -0.5f * d0 * d0 / (1e-15f + s0 * s0)
              -0.5f * d1 * d1 / (1e-15f + s1 * s1);
    w[(size_t)e * kK + k] = __expf(s);
  }
}

// msg[e,c] = sum_k w[e,k]*xg[src[e],k,c]; scatter-add into agg[dst[e],c]
__global__ void k_scatter(const float* __restrict__ xg, const float* __restrict__ we,
                          const int* __restrict__ src, const int* __restrict__ dst,
                          float* __restrict__ agg) {
  int idx = blockIdx.x * blockDim.x + threadIdx.x;
  if (idx >= kE * kC) return;
  int e = idx >> 7, c = idx & 127;
  const float* xr = xg + (size_t)src[e] * kKC + c;
  const float* wr = we + (size_t)e * kK;
  float m = 0.f;
  #pragma unroll
  for (int k = 0; k < kK; ++k) m += wr[k] * xr[k * kC];
  atomicAdd(&agg[(size_t)dst[e] * kC + c], m);
}

// h = agg/deg + hr  (written in-place into hr); BN sum/sumsq per channel
__global__ void k_combine_stats(const float* __restrict__ agg, const float* __restrict__ deg,
                                float* __restrict__ hr, float* __restrict__ stats) {
  __shared__ float ss[kC], sq[kC];
  int tid = threadIdx.x;
  if (tid < kC) { ss[tid] = 0.f; sq[tid] = 0.f; }
  __syncthreads();
  int idx = blockIdx.x * blockDim.x + tid;
  if (idx < kN * kC) {
    int n = idx >> 7, c = idx & 127;
    float h = agg[idx] / fmaxf(deg[n], 1.f) + hr[idx];
    hr[idx] = h;
    atomicAdd(&ss[c], h);
    atomicAdd(&sq[c], h * h);
  }
  __syncthreads();
  if (tid < kC) { atomicAdd(&stats[tid], ss[tid]); atomicAdd(&stats[kC + tid], sq[tid]); }
}

__global__ void k_bn_apply(const float* __restrict__ h, const float* __restrict__ stats,
                           const float* __restrict__ gamma, const float* __restrict__ beta,
                           float* __restrict__ o32, _Float16* __restrict__ o16, int do_relu) {
  int idx = blockIdx.x * blockDim.x + threadIdx.x;
  if (idx >= kN * kC) return;
  int c = idx & 127;
  float mean = stats[c] * (1.f / kN);
  float var  = stats[kC + c] * (1.f / kN) - mean * mean;
  float v = (h[idx] - mean) * rsqrtf(var + 1e-5f) * gamma[c] + beta[c];
  if (do_relu) v = fmaxf(v, 0.f);
  o32[idx] = v;
  o16[idx] = (_Float16)v;
}

// gates [N,4H] order i,f,g,o -> update c,h; emit h in f32 and f16
__global__ void k_lstm_pointwise(const float* __restrict__ gates, float* __restrict__ cst,
                                 float* __restrict__ hst, _Float16* __restrict__ hst16) {
  int idx = blockIdx.x * blockDim.x + threadIdx.x;
  if (idx >= kN * kH) return;
  int n = idx >> 8, j = idx & 255;
  const float* gr = gates + (size_t)n * kH4;
  float gi = gr[j], gf = gr[kH + j], gg = gr[2 * kH + j], go = gr[3 * kH + j];
  float cn = sigmf(gf) * cst[idx] + sigmf(gi) * tanhf(gg);
  float hn = sigmf(go) * tanhf(cn);
  cst[idx] = cn; hst[idx] = hn; hst16[idx] = (_Float16)hn;
}

// alpha[t,n] += h_state[n,:] . att_w(dir slice)
__global__ void k_alpha_dot(const float* __restrict__ hst, const float* __restrict__ aw,
                            float* __restrict__ alpha_t, int n) {
  int i = blockIdx.x * blockDim.x + threadIdx.x;
  if (i >= n) return;
  const float* h = hst + (size_t)i * kH;
  float s = 0.f;
  #pragma unroll 4
  for (int j = 0; j < kH; ++j) s += h[j] * aw[j];
  alpha_t[i] += s;
}

// softmax over T per node, node_rep = sum_t alpha_t * h_list[t]
__global__ void k_softmax_wsum(const float* __restrict__ alpha, const float* __restrict__ attb,
                               const float* __restrict__ hlist, float* __restrict__ nrep) {
  int idx = blockIdx.x * blockDim.x + threadIdx.x;
  if (idx >= kN * kC) return;
  int n = idx >> 7;
  float a[kT], mx = -1e30f;
  #pragma unroll
  for (int t = 0; t < kT; ++t) { a[t] = alpha[(size_t)t * kN + n] + attb[0]; mx = fmaxf(mx, a[t]); }
  float se = 0.f;
  #pragma unroll
  for (int t = 0; t < kT; ++t) { a[t] = __expf(a[t] - mx); se += a[t]; }
  float r = 0.f, inv = 1.f / se;
  #pragma unroll
  for (int t = 0; t < kT; ++t) r += a[t] * inv * hlist[(size_t)t * kN * kC + idx];
  nrep[idx] = r;
}

__global__ void k_count(const int* __restrict__ batch, float* __restrict__ cnt, int n) {
  int i = blockIdx.x * blockDim.x + threadIdx.x;
  if (i < n) atomicAdd(&cnt[batch[i]], 1.f);
}

__global__ void k_pool_scatter(const float* __restrict__ nrep, const int* __restrict__ batch,
                               float* __restrict__ hg) {
  int idx = blockIdx.x * blockDim.x + threadIdx.x;
  if (idx >= kN * kC) return;
  int n = idx >> 7, c = idx & 127;
  atomicAdd(&hg[(size_t)batch[n] * kC + c], nrep[idx]);
}

__global__ void k_pool_div(const float* __restrict__ hg, const float* __restrict__ cnt,
                           _Float16* __restrict__ o16, int n) {
  int idx = blockIdx.x * blockDim.x + threadIdx.x;
  if (idx >= n) return;
  o16[idx] = (_Float16)(hg[idx] / fmaxf(cnt[idx >> 7], 1.f));
}

// LayerNorm over last dim (256) — one block per graph row
__global__ void k_layernorm(const float* __restrict__ z, const float* __restrict__ g,
                            const float* __restrict__ b, float* __restrict__ out) {
  __shared__ float sh[kNOUT];
  int row = blockIdx.x, tid = threadIdx.x;
  float v = z[(size_t)row * kNOUT + tid];
  sh[tid] = v; __syncthreads();
  for (int s = kNOUT / 2; s > 0; s >>= 1) { if (tid < s) sh[tid] += sh[tid + s]; __syncthreads(); }
  float mean = sh[0] * (1.f / kNOUT); __syncthreads();
  float d = v - mean;
  sh[tid] = d * d; __syncthreads();
  for (int s = kNOUT / 2; s > 0; s >>= 1) { if (tid < s) sh[tid] += sh[tid + s]; __syncthreads(); }
  float var = sh[0] * (1.f / kNOUT);
  out[(size_t)row * kNOUT + tid] = d * rsqrtf(var + 1e-5f) * g[tid] + b[tid];
}

// ------------------------------ host side ----------------------------------
static inline void gemm(const _Float16* A, const _Float16* Bt, float* C, const float* bias,
                        int M, int K, int N, int accum, int relu, hipStream_t s) {
  dim3 grid((M + 127) / 128, N / 128);
  gemm_f16_wmma<<<grid, 256, 0, s>>>(A, Bt, C, bias, M, K, N, accum, relu);
}
static inline int cdiv(long long a, long long b) { return (int)((a + b - 1) / b); }

extern "C" void kernel_launch(void* const* d_in, const int* in_sizes, int n_in,
                              void* d_out, int out_size, void* d_ws, size_t ws_size,
                              hipStream_t stream) {
  (void)in_sizes; (void)n_in; (void)out_size; (void)ws_size;
  const float* x      = (const float*)d_in[0];
  const int*   eidx   = (const int*)d_in[1];
  const float* eattr  = (const float*)d_in[2];
  const int*   batch  = (const int*)d_in[3];
  const float* g_w    = (const float*)d_in[4];
  const float* root_w = (const float*)d_in[5];
  const float* bias_w = (const float*)d_in[6];
  const float* mu     = (const float*)d_in[7];
  const float* sigma  = (const float*)d_in[8];
  const float* bng    = (const float*)d_in[9];
  const float* bnb    = (const float*)d_in[10];
  const float* w_ih   = (const float*)d_in[11];
  const float* w_hh   = (const float*)d_in[12];
  const float* b_ih   = (const float*)d_in[13];
  const float* b_hh   = (const float*)d_in[14];
  const float* attw   = (const float*)d_in[15];
  const float* attb   = (const float*)d_in[16];
  const float* p1w    = (const float*)d_in[17];
  const float* p1b    = (const float*)d_in[18];
  const float* p2w    = (const float*)d_in[19];
  const float* p2b    = (const float*)d_in[20];
  const float* lng    = (const float*)d_in[21];
  const float* lnb    = (const float*)d_in[22];
  const int* src = eidx;
  const int* dst = eidx + kE;

  char* wp = (char*)d_ws;
  auto carve = [&](size_t bytes) { void* r = (void*)wp; wp += (bytes + 255) & ~(size_t)255; return r; };
  float*    hlist32 = (float*)   carve((size_t)kT * kN * kC * 4);
  _Float16* hlist16 = (_Float16*)carve((size_t)kT * kN * kC * 2);
  float*    big     = (float*)   carve((size_t)kN * kH4 * 4);      // xg / LSTM gates (shared)
  float*    agg     = (float*)   carve((size_t)kN * kC * 4);
  float*    hr      = (float*)   carve((size_t)kN * kC * 4);       // h@root+b, then node_rep
  float*    wedge   = (float*)   carve((size_t)kE * kK * 4);
  float*    deg     = (float*)   carve((size_t)kN * 4);
  float*    stats   = (float*)   carve(2 * kC * 4);
  // all B operands stored pre-transposed as [N][K] f16
  _Float16* g16     = (_Float16*)carve((size_t)kL * kKC * kC * 2);   // [1024][128] / layer
  _Float16* root16  = (_Float16*)carve((size_t)kL * kC * kC * 2);    // [128][128]  / layer
  _Float16* wih16   = (_Float16*)carve((size_t)2 * kH4 * kC * 2);    // [1024][128] / dir
  _Float16* whh16   = (_Float16*)carve((size_t)2 * kH4 * kH * 2);    // [1024][256] / dir
  _Float16* p116    = (_Float16*)carve((size_t)kNHID * kC * 2);      // [512][128]
  _Float16* p216    = (_Float16*)carve((size_t)kNOUT * kNHID * 2);   // [256][512]
  float*    bsum    = (float*)   carve(2 * kH4 * 4);
  float*    hst     = (float*)   carve((size_t)kN * kH * 4);
  float*    cst     = (float*)   carve((size_t)kN * kH * 4);
  _Float16* hst16   = (_Float16*)carve((size_t)kN * kH * 2);
  float*    alpha   = (float*)   carve((size_t)kT * kN * 4);
  float*    hgraph  = (float*)   carve((size_t)kB * kC * 4);
  float*    counts  = (float*)   carve((size_t)kB * 4);
  _Float16* hg16    = (_Float16*)carve((size_t)kB * kC * 2);
  float*    z1      = (float*)   carve((size_t)kB * kNHID * 4);
  _Float16* z116    = (_Float16*)carve((size_t)kB * kNHID * 2);
  float*    z2      = (float*)   carve((size_t)kB * kNOUT * 4);

  // ---- weight conversions (f32 -> f16), stored as B^T = [N][K] ----
  for (int i = 0; i < kL; ++i) {
    k_transpose_f16<<<cdiv(kC * kKC, 256), 256, 0, stream>>>(
        g_w + (size_t)i * kC * kKC, g16 + (size_t)i * kKC * kC, kC, kKC);
    k_transpose_f16<<<cdiv(kC * kC, 256), 256, 0, stream>>>(
        root_w + (size_t)i * kC * kC, root16 + (size_t)i * kC * kC, kC, kC);
  }
  // w_ih [4H,C] and w_hh [4H,H] are already [N][K] for x@W^T: plain convert
  k_f32_to_f16<<<cdiv(2 * kH4 * kC, 256), 256, 0, stream>>>(w_ih, wih16, 2 * kH4 * kC);
  k_f32_to_f16<<<cdiv(2 * kH4 * kH, 256), 256, 0, stream>>>(w_hh, whh16, 2 * kH4 * kH);
  k_transpose_f16<<<cdiv(kC * kNHID, 256), 256, 0, stream>>>(p1w, p116, kC, kNHID);
  k_transpose_f16<<<cdiv(kNHID * kNOUT, 256), 256, 0, stream>>>(p2w, p216, kNHID, kNOUT);
  k_bias_sum<<<cdiv(2 * kH4, 256), 256, 0, stream>>>(b_ih, b_hh, bsum, 2 * kH4);

  // ---- h_list[0] = x ----
  hipMemcpyAsync(hlist32, x, (size_t)kN * kC * 4, hipMemcpyDeviceToDevice, stream);
  k_f32_to_f16<<<cdiv(kN * kC, 256), 256, 0, stream>>>(x, hlist16, kN * kC);

  // ---- degree ----
  hipMemsetAsync(deg, 0, (size_t)kN * 4, stream);
  k_deg<<<cdiv(kE, 256), 256, 0, stream>>>(dst, deg, kE);

  // ---- GMMConv layers ----
  for (int i = 0; i < kL; ++i) {
    const _Float16* hin = hlist16 + (size_t)i * kN * kC;
    gemm(hin, g16 + (size_t)i * kKC * kC, big, nullptr, kN, kC, kKC, 0, 0, stream);   // xg
    gemm(hin, root16 + (size_t)i * kC * kC, hr, bias_w + i * kC, kN, kC, kC, 0, 0, stream);
    k_edge_w<<<cdiv(kE, 256), 256, 0, stream>>>(eattr, mu + i * kK * kD,
                                                sigma + i * kK * kD, wedge, kE);
    hipMemsetAsync(agg, 0, (size_t)kN * kC * 4, stream);
    k_scatter<<<cdiv((long long)kE * kC, 256), 256, 0, stream>>>(big, wedge, src, dst, agg);
    hipMemsetAsync(stats, 0, 2 * kC * 4, stream);
    k_combine_stats<<<cdiv(kN * kC, 256), 256, 0, stream>>>(agg, deg, hr, stats);
    k_bn_apply<<<cdiv(kN * kC, 256), 256, 0, stream>>>(
        hr, stats, bng + i * kC, bnb + i * kC,
        hlist32 + (size_t)(i + 1) * kN * kC, hlist16 + (size_t)(i + 1) * kN * kC,
        (i < kL - 1) ? 1 : 0);
  }

  // ---- BiLSTM JK, alpha accumulated on the fly (no [T,N,2H] buffer) ----
  hipMemsetAsync(alpha, 0, (size_t)kT * kN * 4, stream);
  for (int d = 0; d < 2; ++d) {
    hipMemsetAsync(hst, 0, (size_t)kN * kH * 4, stream);
    hipMemsetAsync(cst, 0, (size_t)kN * kH * 4, stream);
    hipMemsetAsync(hst16, 0, (size_t)kN * kH * 2, stream);
    for (int s = 0; s < kT; ++s) {
      int t = d ? (kT - 1 - s) : s;
      gemm(hlist16 + (size_t)t * kN * kC, wih16 + (size_t)d * kH4 * kC, big,
           bsum + d * kH4, kN, kC, kH4, 0, 0, stream);                    // x @ w_ih^T + b
      gemm(hst16, whh16 + (size_t)d * kH4 * kH, big, nullptr,
           kN, kH, kH4, 1, 0, stream);                                    // += h @ w_hh^T
      k_lstm_pointwise<<<cdiv(kN * kH, 256), 256, 0, stream>>>(big, cst, hst, hst16);
      k_alpha_dot<<<cdiv(kN, 256), 256, 0, stream>>>(hst, attw + d * kH,
                                                     alpha + (size_t)t * kN, kN);
    }
  }

  // ---- attention softmax + weighted sum -> node_rep (reuse hr) ----
  float* noderep = hr;
  k_softmax_wsum<<<cdiv(kN * kC, 256), 256, 0, stream>>>(alpha, attb, hlist32, noderep);

  // ---- global mean pool ----
  hipMemsetAsync(hgraph, 0, (size_t)kB * kC * 4, stream);
  hipMemsetAsync(counts, 0, (size_t)kB * 4, stream);
  k_count<<<cdiv(kN, 256), 256, 0, stream>>>(batch, counts, kN);
  k_pool_scatter<<<cdiv(kN * kC, 256), 256, 0, stream>>>(noderep, batch, hgraph);
  k_pool_div<<<cdiv(kB * kC, 256), 256, 0, stream>>>(hgraph, counts, hg16, kB * kC);

  // ---- MLP + LayerNorm ----
  gemm(hg16, p116, z1, p1b, kB, kC, kNHID, 0, 1, stream);
  k_f32_to_f16<<<cdiv(kB * kNHID, 256), 256, 0, stream>>>(z1, z116, kB * kNHID);
  gemm(z116, p216, z2, p2b, kB, kNHID, kNOUT, 0, 0, stream);
  k_layernorm<<<kB, kNOUT, 0, stream>>>(z2, lng, lnb, (float*)d_out);
}
